// MultiHeadAttention_50079318671839
// MI455X (gfx1250) — matrix operand order repaired and checked by reading, not compiled, not run on hardware
//
#include <hip/hip_runtime.h>
#include <hip/hip_bf16.h>
#include <math.h>

typedef unsigned short u16;
typedef unsigned int   u32;
typedef __attribute__((ext_vector_type(16))) __bf16 v16bf;
typedef __attribute__((ext_vector_type(8)))  float  v8f;
typedef __attribute__((ext_vector_type(4)))  int    i32x4;

#define SEQ    2048
#define BATCH  2
#define NHEADS 16
#define DKV    64
#define DMODEL 1024
#define MROWS  (BATCH * SEQ)   // 4096
#define BIASW  4095            // relpos in [-2047, 2047]

// ---------- CDNA5 async global->LDS copy (ASYNCcnt path) ----------

#if defined(__gfx1250__) && __has_builtin(__builtin_amdgcn_global_load_async_to_lds_b128)
#define HAVE_ASYNC_LDS 1
#else
#define HAVE_ASYNC_LDS 0
#endif

#if HAVE_ASYNC_LDS
typedef __attribute__((address_space(1))) void* gas_t;
typedef __attribute__((address_space(3))) void* las_t;
typedef __attribute__((address_space(1))) i32x4* gv4_t;   // AS1 int4*
typedef __attribute__((address_space(3))) i32x4* lv4_t;   // AS3 int4*
__device__ __forceinline__ void cp_async16(u16* lds, const u16* g) {
  __builtin_amdgcn_global_load_async_to_lds_b128(
      (gv4_t)(gas_t)g, (lv4_t)(las_t)lds, 0, 0);
}
__device__ __forceinline__ void wait_async0() {
#if __has_builtin(__builtin_amdgcn_s_wait_asynccnt)
  __builtin_amdgcn_s_wait_asynccnt(0);
#else
  asm volatile("s_wait_asynccnt 0x0" ::: "memory");
#endif
}
#endif

// ---------- helpers ----------

__device__ __forceinline__ u16 f2bf(float f) {
  u32 u = __float_as_uint(f);
  u32 r = u + 0x7FFFu + ((u >> 16) & 1u);   // round-to-nearest-even
  return (u16)(r >> 16);
}

union AF { v16bf v; uint4 q[2]; };  // 32-byte WMMA 16-bit operand fragment

__device__ __forceinline__ v8f wmma_bf16(v16bf a, v16bf b, v8f c) {
  // (neg_a, A, neg_b, B, c_mod, C, reuse_a, reuse_b)
  return __builtin_amdgcn_wmma_f32_16x16x32_bf16(false, a, false, b, (short)0, c,
                                                 false, false);
}

// ---------- prep kernels ----------

// hidden f32 -> bf16, 4 elems/thread
__global__ void k_prep_convert(const float* __restrict__ H, u16* __restrict__ Hb) {
  int i = (blockIdx.x * blockDim.x + threadIdx.x) * 4;
  float4 f = *(const float4*)(H + i);
  uint2 o;
  o.x = (u32)f2bf(f.x) | ((u32)f2bf(f.y) << 16);
  o.y = (u32)f2bf(f.z) | ((u32)f2bf(f.w) << 16);
  *(uint2*)(Hb + i) = o;
}

// W[k][n] f32 -> Wt[n][k] bf16 (z picks Wq/Wk/Wv/Wo)
__global__ void k_prep_transpose(const float* __restrict__ Wq, const float* __restrict__ Wk,
                                 const float* __restrict__ Wv, const float* __restrict__ Wo,
                                 u16* __restrict__ Wt) {
  const float* W = (blockIdx.z == 0) ? Wq : (blockIdx.z == 1) ? Wk
                   : (blockIdx.z == 2) ? Wv : Wo;
  u16* O = Wt + (size_t)blockIdx.z * DMODEL * DMODEL;
  int idx = blockIdx.x * blockDim.x + threadIdx.x;     // 0..1M-1
  int k = idx >> 10, n = idx & 1023;                   // coalesced read along n
  O[(size_t)n * DMODEL + k] = f2bf(W[idx]);
}

// full T5 bidirectional bias row per head: bias[h][relpos + 2047]
__global__ void k_prep_bias(const float* __restrict__ table, float* __restrict__ bias) {
  int i = blockIdx.x * blockDim.x + threadIdx.x;
  if (i >= NHEADS * BIASW) return;
  int h = i / BIASW, idx = i % BIASW;
  int rel = idx - (SEQ - 1);                 // k - q
  int bkt = (rel > 0) ? 16 : 0;              // NUM_BUCKETS/2
  int rp = rel < 0 ? -rel : rel;
  if (rp < 8) {
    bkt += rp;                               // max_exact = 8
  } else {
    int rl = 8 + (int)(logf((float)rp * 0.125f) * (8.0f / logf(16.0f)));
    bkt += (rl < 15 ? rl : 15);
  }
  bias[i] = table[bkt * NHEADS + h];
}

// ---------- 128x128 bf16 GEMM mainloop (K=1024), shared by both GEMMs ----------
// A: [M][1024] bf16 row-major, B: [N][1024] bf16 row-major (pre-transposed weight).
// 8 waves, wave tile 64(M) x 32(N) = 4x2 accs of 16x16.
// Explicit LDS double buffering; copies go over the CDNA5 async-to-LDS path
// (ASYNCcnt) when available so tile t+1 streams in under tile t's WMMAs.

__device__ __forceinline__ void gemm128x128(const u16* __restrict__ A,
                                            const u16* __restrict__ B,
                                            u16* sA, u16* sB,   // 2 * 128*32 each
                                            int rowbase, int colbase,
                                            v8f acc[4][2]) {
  const int tid = threadIdx.x;
  const int wid = tid >> 5, lane = tid & 31;
  const int lr = lane & 15, hf = lane >> 4;
  const int wm = wid >> 2, wn = wid & 3;

  for (int i = 0; i < 4; i++)
    for (int j = 0; j < 2; j++)
      for (int e = 0; e < 8; e++) acc[i][j][e] = 0.f;

  const int lm  = tid >> 1;          // 0..127 tile row
  const int seg = (tid & 1) * 16;    // half of the 32-wide K slab

  auto issue = [&](int k0, int buf) {
    u16* dA = sA + buf * (128 * 32) + lm * 32 + seg;
    u16* dB = sB + buf * (128 * 32) + lm * 32 + seg;
    const u16* ga = A + (size_t)(rowbase + lm) * DMODEL + k0 + seg;
    const u16* gb = B + (size_t)(colbase + lm) * DMODEL + k0 + seg;
#if HAVE_ASYNC_LDS
    cp_async16(dA,     ga);
    cp_async16(dA + 8, ga + 8);
    cp_async16(dB,     gb);
    cp_async16(dB + 8, gb + 8);
#else
    ((uint4*)dA)[0] = ((const uint4*)ga)[0];
    ((uint4*)dA)[1] = ((const uint4*)ga)[1];
    ((uint4*)dB)[0] = ((const uint4*)gb)[0];
    ((uint4*)dB)[1] = ((const uint4*)gb)[1];
#endif
  };
  auto fence = [&]() {
#if HAVE_ASYNC_LDS
    wait_async0();
#endif
    __syncthreads();
  };

  issue(0, 0);
  fence();

  for (int k0 = 0; k0 < DMODEL; k0 += 32) {
    const int cur = (k0 >> 5) & 1;
    if (k0 + 32 < DMODEL) issue(k0 + 32, cur ^ 1);   // prefetch next tile

    const u16* cA = sA + cur * (128 * 32);
    const u16* cB = sB + cur * (128 * 32);
    AF a[4], b[2];
    // A fragment (16x32 bf16): lanes 0-15 rows M, elems 0..7 = K hf*8..hf*8+7,
    // elems 8..15 = K 16+hf*8.. (per ISA 7.12.2)
    for (int i = 0; i < 4; i++) {
      const u16* p = cA + (wm * 64 + i * 16 + lr) * 32 + hf * 8;
      a[i].q[0] = *(const uint4*)p;
      a[i].q[1] = *(const uint4*)(p + 16);
    }
    // B fragment (32x16 bf16): col N=lane%16, same K order as A:
    // elems 0..7 = K hf*8.., elems 8..15 = K 16+hf*8..
    for (int j = 0; j < 2; j++) {
      const u16* p = cB + (wn * 32 + j * 16 + lr) * 32 + hf * 8;
      b[j].q[0] = *(const uint4*)p;
      b[j].q[1] = *(const uint4*)(p + 16);
    }
    for (int i = 0; i < 4; i++)
      for (int j = 0; j < 2; j++)
        acc[i][j] = wmma_bf16(a[i].v, b[j].v, acc[i][j]);

    fence();   // own async copies landed + all waves done reading `cur`
  }
}

// ---------- QKV projection: z=0 Q, z=1 K (both [b][h][s][d]), z=2 V transposed [b][h][d][s] ----------

__global__ void __launch_bounds__(256, 1)
k_qkv_gemm(const u16* __restrict__ Hb, const u16* __restrict__ Wt,
           u16* __restrict__ Qb, u16* __restrict__ Kb, u16* __restrict__ Vt) {
  __shared__ __align__(16) u16 sA[2 * 128 * 32];
  __shared__ __align__(16) u16 sB[2 * 128 * 32];
  const int z = blockIdx.z;
  const u16* B = Wt + (size_t)z * DMODEL * DMODEL;
  int rowbase = blockIdx.y * 128, colbase = blockIdx.x * 128;

  v8f acc[4][2];
  gemm128x128(Hb, B, sA, sB, rowbase, colbase, acc);

  const int tid = threadIdx.x, wid = tid >> 5, lane = tid & 31;
  const int lr = lane & 15, hf = lane >> 4;
  const int wm = wid >> 2, wn = wid & 3;
  for (int i = 0; i < 4; i++)
    for (int j = 0; j < 2; j++) {
      int mb = rowbase + wm * 64 + i * 16 + 8 * hf;  // C layout: M = r + 8*hf
      int nc = colbase + wn * 32 + j * 16 + lr;      // N = lane%16
      int hh = nc >> 6, dd = nc & 63;
      for (int r = 0; r < 8; r++) {
        int grow = mb + r;
        int bb = grow >> 11, ss = grow & 2047;
        u16 val = f2bf(acc[i][j][r]);
        if (z == 0)
          Qb[(((size_t)bb * NHEADS + hh) * SEQ + ss) * DKV + dd] = val;
        else if (z == 1)
          Kb[(((size_t)bb * NHEADS + hh) * SEQ + ss) * DKV + dd] = val;
        else
          Vt[(((size_t)bb * NHEADS + hh) * DKV + dd) * SEQ + ss] = val;
      }
    }
}

// ---------- flash attention: one block = (b,h) x 128 q-rows; wave owns 16 rows ----------

__global__ void __launch_bounds__(256, 1)
k_attention(const u16* __restrict__ Qb, const u16* __restrict__ Kb,
            const u16* __restrict__ Vt, const float* __restrict__ bias,
            u16* __restrict__ AO) {
  __shared__ __align__(16) u16 sP[8][16 * 64];  // per-wave P relayout scratch
  const int tid = threadIdx.x, wid = tid >> 5, lane = tid & 31;
  const int lr = lane & 15, hf = lane >> 4;
  const int bh = blockIdx.y, b = bh >> 4, h = bh & 15;
  const int q0 = blockIdx.x * 128 + wid * 16;

  // Q fragments (A layout), 16 rows x 64 d -> two 16x32 slices
  const u16* qp = Qb + ((size_t)bh * SEQ + q0) * DKV;
  AF qf[2];
  for (int s = 0; s < 2; s++) {
    const u16* p = qp + lr * DKV + s * 32 + hf * 8;
    qf[s].q[0] = *(const uint4*)p;
    qf[s].q[1] = *(const uint4*)(p + 16);
  }

  v8f o[4];
  for (int jd = 0; jd < 4; jd++)
    for (int e = 0; e < 8; e++) o[jd][e] = 0.f;
  float mrow[8], lrw[8];
  for (int r = 0; r < 8; r++) { mrow[r] = -3.0e38f; lrw[r] = 0.f; }

  // bias[h][key - q + 2047]; per-lane rows are q0 + 8*hf + r
  const float* brow = bias + h * BIASW + (SEQ - 1) - (q0 + 8 * hf);
  u16* mp = sP[wid];

  const u16* kbase = Kb + (size_t)bh * SEQ * DKV;
  const u16* vbase = Vt + (size_t)bh * DKV * SEQ;

  for (int kt = 0; kt < SEQ / 64; kt++) {
    const u16* kp = kbase + (size_t)kt * 64 * DKV;

    // prefetch next K/V 64-key tiles (global_prefetch_b8) under this tile's math
    if (kt + 1 < SEQ / 64) {
      const u16* kn = kp + 64 * DKV;                 // next K tile: 8 KB contiguous
      __builtin_prefetch(kn + (size_t)lane * 128, 0, 0);
      __builtin_prefetch(kn + (size_t)lane * 128 + 64, 0, 0);
      const u16* vn = vbase + (kt + 1) * 64;         // next V tile: 64 rows x 128 B
      __builtin_prefetch(vn + (size_t)(2 * lane) * SEQ, 0, 0);
      __builtin_prefetch(vn + (size_t)(2 * lane + 1) * SEQ, 0, 0);
    }

    v8f sacc[4];
    for (int j = 0; j < 4; j++) {
      // B fragments of K^T: col = key (row-major K[key][d] => contiguous along d)
      const u16* kcol = kp + (j * 16 + lr) * DKV;
      AF b0, b1;
      b0.q[0] = *(const uint4*)(kcol + hf * 8);
      b0.q[1] = *(const uint4*)(kcol + hf * 8 + 16);
      b1.q[0] = *(const uint4*)(kcol + 32 + hf * 8);
      b1.q[1] = *(const uint4*)(kcol + 32 + hf * 8 + 16);
      v8f a;
      for (int e = 0; e < 8; e++) a[e] = 0.f;
      a = wmma_bf16(qf[0].v, b0.v, a);
      a = wmma_bf16(qf[1].v, b1.v, a);
      int key = kt * 64 + j * 16 + lr;
      for (int r = 0; r < 8; r++) a[r] += brow[key - r];  // additive T5 bias
      sacc[j] = a;
    }

    // online softmax: row reductions across the 16 lanes of each half-group
    float rmax[8];
    for (int r = 0; r < 8; r++)
      rmax[r] = fmaxf(fmaxf(sacc[0][r], sacc[1][r]), fmaxf(sacc[2][r], sacc[3][r]));
    for (int msk = 1; msk < 16; msk <<= 1)
      for (int r = 0; r < 8; r++)
        rmax[r] = fmaxf(rmax[r], __shfl_xor(rmax[r], msk, 32));
    float scl[8], rsum[8];
    for (int r = 0; r < 8; r++) {
      float mn = fmaxf(mrow[r], rmax[r]);
      scl[r] = __expf(mrow[r] - mn);
      mrow[r] = mn;
      rsum[r] = 0.f;
    }
    for (int j = 0; j < 4; j++)
      for (int r = 0; r < 8; r++) {
        float p = __expf(sacc[j][r] - mrow[r]);
        sacc[j][r] = p;
        rsum[r] += p;
      }
    for (int msk = 1; msk < 16; msk <<= 1)
      for (int r = 0; r < 8; r++) rsum[r] += __shfl_xor(rsum[r], msk, 32);
    for (int r = 0; r < 8; r++) lrw[r] = lrw[r] * scl[r] + rsum[r];
    for (int jd = 0; jd < 4; jd++)
      for (int r = 0; r < 8; r++) o[jd][r] *= scl[r];

    // P: C layout -> row-major bf16 in per-wave LDS, then reload as A fragments
    for (int j = 0; j < 4; j++)
      for (int r = 0; r < 8; r++)
        mp[(8 * hf + r) * 64 + j * 16 + lr] = f2bf(sacc[j][r]);
    asm volatile("s_wait_dscnt 0" ::: "memory");  // wave-local LDS RAW fence

    AF pf[2];
    for (int sl = 0; sl < 2; sl++) {
      const u16* p = mp + lr * 64 + sl * 32 + hf * 8;
      pf[sl].q[0] = *(const uint4*)p;
      pf[sl].q[1] = *(const uint4*)(p + 16);
    }

    // O += P * V ; Vt[b][h][d][s] makes B fragments contiguous along keys
    const u16* vb = vbase + kt * 64;
    for (int jd = 0; jd < 4; jd++)
      for (int sl = 0; sl < 2; sl++) {
        AF vf;
        const u16* p = vb + (size_t)(jd * 16 + lr) * SEQ + sl * 32 + hf * 8;
        vf.q[0] = *(const uint4*)p;
        vf.q[1] = *(const uint4*)(p + 16);
        o[jd] = wmma_bf16(pf[sl].v, vf.v, o[jd]);
      }
  }

  // normalize and write AO[b][s][h*64+d] as bf16 for the output projection
  float inv[8];
  for (int r = 0; r < 8; r++) inv[r] = 1.0f / lrw[r];
  for (int jd = 0; jd < 4; jd++) {
    int d = jd * 16 + lr;
    for (int r = 0; r < 8; r++) {
      int sg = q0 + 8 * hf + r;
      AO[((size_t)b * SEQ + sg) * DMODEL + h * DKV + d] = f2bf(o[jd][r] * inv[r]);
    }
  }
}

// ---------- output projection -> f32 d_out ----------

__global__ void __launch_bounds__(256, 1)
k_out_gemm(const u16* __restrict__ AO, const u16* __restrict__ Wot,
           float* __restrict__ Out) {
  __shared__ __align__(16) u16 sA[2 * 128 * 32];
  __shared__ __align__(16) u16 sB[2 * 128 * 32];
  int rowbase = blockIdx.y * 128, colbase = blockIdx.x * 128;

  v8f acc[4][2];
  gemm128x128(AO, Wot, sA, sB, rowbase, colbase, acc);

  const int tid = threadIdx.x, wid = tid >> 5, lane = tid & 31;
  const int lr = lane & 15, hf = lane >> 4;
  const int wm = wid >> 2, wn = wid & 3;
  for (int i = 0; i < 4; i++)
    for (int j = 0; j < 2; j++) {
      int mb = rowbase + wm * 64 + i * 16 + 8 * hf;
      int nc = colbase + wn * 32 + j * 16 + lr;
      for (int r = 0; r < 8; r++)
        Out[(size_t)(mb + r) * DMODEL + nc] = acc[i][j][r];
    }
}

// ---------- host ----------

extern "C" void kernel_launch(void* const* d_in, const int* in_sizes, int n_in,
                              void* d_out, int out_size, void* d_ws, size_t ws_size,
                              hipStream_t stream) {
  const float* H  = (const float*)d_in[0];
  const float* Wq = (const float*)d_in[1];
  const float* Wk = (const float*)d_in[2];
  const float* Wv = (const float*)d_in[3];
  const float* Wo = (const float*)d_in[4];
  const float* tb = (const float*)d_in[5];

  char* ws = (char*)d_ws;
  size_t off = 0;
  auto alloc = [&](size_t bytes) -> void* {
    void* p = ws + off;
    off += (bytes + 255) & ~(size_t)255;
    return p;
  };
  u16*   Hb   = (u16*)alloc((size_t)MROWS * DMODEL * 2);
  u16*   Wt   = (u16*)alloc((size_t)4 * DMODEL * DMODEL * 2);
  u16*   Qb   = (u16*)alloc((size_t)BATCH * NHEADS * SEQ * DKV * 2);
  u16*   Kb   = (u16*)alloc((size_t)BATCH * NHEADS * SEQ * DKV * 2);
  u16*   Vt   = (u16*)alloc((size_t)BATCH * NHEADS * SEQ * DKV * 2);
  u16*   AO   = (u16*)alloc((size_t)MROWS * DMODEL * 2);
  float* bias = (float*)alloc((size_t)NHEADS * BIASW * 4);

  k_prep_convert<<<MROWS * DMODEL / (256 * 4), 256, 0, stream>>>(H, Hb);
  k_prep_transpose<<<dim3(DMODEL * DMODEL / 256, 1, 4), 256, 0, stream>>>(Wq, Wk, Wv, Wo, Wt);
  k_prep_bias<<<(NHEADS * BIASW + 255) / 256, 256, 0, stream>>>(tb, bias);

  k_qkv_gemm<<<dim3(DMODEL / 128, MROWS / 128, 3), 256, 0, stream>>>(Hb, Wt, Qb, Kb, Vt);
  k_attention<<<dim3(SEQ / 128, BATCH * NHEADS), 256, 0, stream>>>(Qb, Kb, Vt, bias, AO);
  k_out_gemm<<<dim3(DMODEL / 128, MROWS / 128), 256, 0, stream>>>(
      AO, Wt + (size_t)3 * DMODEL * DMODEL, (float*)d_out);
}
